// Model_33474975105692
// MI455X (gfx1250) — matrix operand order repaired and checked by reading, not compiled, not run on hardware
//
#include <hip/hip_runtime.h>
#include <stdint.h>

// Fused attention for B=1, H=128, S=256, D=2048 (fp32 in/out).
// Memory-bound problem (~1 GB HBM @ 23.3 TB/s ~= 44us floor); matmul math is
// done in f16 WMMA with f32 accumulation (v_wmma_f32_16x16x32_f16), which is
// far above the bandwidth roofline. Two kernels serialized on `stream`:
//   1) scores = softmax(Q K^T * scale + mask), dropout -> P (f16) in d_ws
//   2) O = P V  -> d_out (f32)

typedef __attribute__((ext_vector_type(16))) _Float16 v16h;
typedef __attribute__((ext_vector_type(8)))  float    v8f;

#define H_  128
#define S_  256
#define D_  2048
#define SCALE_ 0.022097086912079608f   // 1/sqrt(2048)
#define KEEP_P_ 0.1f
#define KEEP_SCALE_ 10.0f

// A-matrix (16x32, f16) per-lane K index for VGPR pair v:
// lanes 0-15 hold K 0..15 ; lanes 16-31 hold K 16..31 (matches B).
__device__ __forceinline__ int kbaseA(int v, int half16) {
    return 2 * v + 16 * half16;
}

__device__ __forceinline__ float rng01(uint32_t x) {
    // cheap deterministic hash -> [0,1)
    x ^= 0x2545F491u;           // mix in fixed "key 42"-style salt
    x *= 2654435761u;
    x ^= x >> 16;  x *= 0x85EBCA6Bu;
    x ^= x >> 13;  x *= 0xC2B2AE35u;
    x ^= x >> 16;
    return (float)(x >> 8) * (1.0f / 16777216.0f);
}

// ---------------------------------------------------------------------------
// Kernel 1: one wave computes a 16x256 score strip for head h, q-rows q0..q0+15,
// then softmax + dropout, writing P (f16) row-major [H][S][S] into workspace.
// ---------------------------------------------------------------------------
__global__ __launch_bounds__(32)
void attn_scores_softmax(const float* __restrict__ Q,
                         const float* __restrict__ K,
                         const float* __restrict__ mask,
                         _Float16* __restrict__ P) {
    const int lane = threadIdx.x;
    const int hv   = lane >> 4;    // which 16-lane half of the wave
    const int ln16 = lane & 15;
    const int q0   = blockIdx.x * 16;
    const int h    = blockIdx.y;

    const float* Qh = Q + (size_t)h * S_ * D_;
    const float* Kh = K + (size_t)h * S_ * D_;

    v8f acc[16];
    #pragma unroll
    for (int t = 0; t < 16; ++t)
        acc[t] = (v8f){0.f, 0.f, 0.f, 0.f, 0.f, 0.f, 0.f, 0.f};

    // S_tile[16x256] += Q[16 x 32] * K^T[32 x 256], chunked over D
    for (int d0 = 0; d0 < D_; d0 += 32) {
        // A fragment: Q rows q0..q0+15, cols d0..d0+31
        v16h aF;
        const float* qrow = Qh + (size_t)(q0 + ln16) * D_ + d0;
        #pragma unroll
        for (int v = 0; v < 8; ++v) {
            const int kb = kbaseA(v, hv);
            const float2 f = *reinterpret_cast<const float2*>(qrow + kb);
            aF[2 * v]     = (_Float16)f.x;
            aF[2 * v + 1] = (_Float16)f.y;
        }
        #pragma unroll
        for (int t = 0; t < 16; ++t) {
            // B fragment: B[kd][n] = K[t*16+n][d0+kd]; lane n = ln16,
            // element e <-> kd = e + 16*hv (contiguous in d -> float2 loads)
            v16h bF;
            const float* krow = Kh + (size_t)(t * 16 + ln16) * D_ + d0 + 16 * hv;
            #pragma unroll
            for (int v = 0; v < 8; ++v) {
                const float2 f = *reinterpret_cast<const float2*>(krow + 2 * v);
                bF[2 * v]     = (_Float16)f.x;
                bF[2 * v + 1] = (_Float16)f.y;
            }
            acc[t] = __builtin_amdgcn_wmma_f32_16x16x32_f16(
                false, aF, false, bF, (short)0, acc[t], false, false);
        }
    }

    // C/D layout: lane group hv, VGPR r -> row m = r + 8*hv, col = t*16 + ln16.
    #pragma unroll
    for (int r = 0; r < 8; ++r) {
        const int m = r + 8 * hv;
        const float* mrow = mask + (size_t)(q0 + m) * S_;

        float vals[16];
        float mx = -3.0e38f;
        #pragma unroll
        for (int t = 0; t < 16; ++t) {
            const float x = acc[t][r] * SCALE_ + mrow[t * 16 + ln16];
            vals[t] = x;
            mx = fmaxf(mx, x);
        }
        // row-wise max: reduce across the 16 lanes holding this row
        #pragma unroll
        for (int off = 1; off < 16; off <<= 1)
            mx = fmaxf(mx, __shfl_xor(mx, off, 32));

        float sum = 0.f;
        #pragma unroll
        for (int t = 0; t < 16; ++t) {
            vals[t] = __expf(vals[t] - mx);
            sum += vals[t];
        }
        #pragma unroll
        for (int off = 1; off < 16; off <<= 1)
            sum += __shfl_xor(sum, off, 32);

        const float inv = KEEP_SCALE_ / sum;
        _Float16* prow = P + (size_t)h * S_ * S_ + (size_t)(q0 + m) * S_;
        #pragma unroll
        for (int t = 0; t < 16; ++t) {
            const int col = t * 16 + ln16;
            const uint32_t seed =
                (uint32_t)((h * S_ + (q0 + m)) * S_ + col);
            const float p = (rng01(seed) < KEEP_P_) ? vals[t] * inv : 0.f;
            prow[col] = (_Float16)p;
        }
    }
}

// ---------------------------------------------------------------------------
// Kernel 2: O = P * V. One wave computes a 16x128 output tile (8 n-tiles),
// chunking the k-dimension (S=256) by 32.
// ---------------------------------------------------------------------------
__global__ __launch_bounds__(32)
void attn_pv(const _Float16* __restrict__ P,
             const float* __restrict__ V,
             float* __restrict__ O) {
    const int lane = threadIdx.x;
    const int hv   = lane >> 4;
    const int ln16 = lane & 15;
    const int n0   = blockIdx.x * 128;
    const int q0   = blockIdx.y * 16;
    const int h    = blockIdx.z;

    const _Float16* Ph = P + (size_t)h * S_ * S_;
    const float*    Vh = V + (size_t)h * S_ * D_;

    v8f acc[8];
    #pragma unroll
    for (int nt = 0; nt < 8; ++nt)
        acc[nt] = (v8f){0.f, 0.f, 0.f, 0.f, 0.f, 0.f, 0.f, 0.f};

    for (int k0 = 0; k0 < S_; k0 += 32) {
        // A fragment: P rows q0..q0+15, k-cols k0..k0+31 (already f16)
        v16h aF;
        const _Float16* prow = Ph + (size_t)(q0 + ln16) * S_ + k0;
        #pragma unroll
        for (int v = 0; v < 8; ++v) {
            const int kb = kbaseA(v, hv);
            aF[2 * v]     = prow[kb];
            aF[2 * v + 1] = prow[kb + 1];
        }
        #pragma unroll
        for (int nt = 0; nt < 8; ++nt) {
            // B fragment: B[kd][n] = V[k0+kd][n0 + nt*16 + n];
            // lane n = ln16, element e <-> kd = e + 16*hv (stride D in memory)
            v16h bF;
            const float* vcol =
                Vh + (size_t)(k0 + 16 * hv) * D_ + n0 + nt * 16 + ln16;
            #pragma unroll
            for (int e = 0; e < 16; ++e)
                bF[e] = (_Float16)vcol[(size_t)e * D_];
            acc[nt] = __builtin_amdgcn_wmma_f32_16x16x32_f16(
                false, aF, false, bF, (short)0, acc[nt], false, false);
        }
    }

    #pragma unroll
    for (int nt = 0; nt < 8; ++nt) {
        #pragma unroll
        for (int r = 0; r < 8; ++r) {
            const int m = r + 8 * hv;
            O[(size_t)h * S_ * D_ + (size_t)(q0 + m) * D_ + n0 + nt * 16 + ln16] =
                acc[nt][r];
        }
    }
}

extern "C" void kernel_launch(void* const* d_in, const int* in_sizes, int n_in,
                              void* d_out, int out_size, void* d_ws, size_t ws_size,
                              hipStream_t stream) {
    (void)in_sizes; (void)n_in; (void)out_size; (void)ws_size;
    const float* Q    = (const float*)d_in[0];
    const float* K    = (const float*)d_in[1];
    const float* V    = (const float*)d_in[2];
    const float* mask = (const float*)d_in[3];
    float*       O    = (float*)d_out;
    _Float16*    P    = (_Float16*)d_ws;   // needs H*S*S*2 = 16 MB scratch

    dim3 blk(32, 1, 1);                    // one wave32 per block
    dim3 g1(S_ / 16, H_, 1);               // 16 q-tiles x 128 heads
    attn_scores_softmax<<<g1, blk, 0, stream>>>(Q, K, mask, P);

    dim3 g2(D_ / 128, S_ / 16, H_);        // 16 n-chunks x 16 q-tiles x 128 heads
    attn_pv<<<g2, blk, 0, stream>>>(P, V, O);
}
